// DepthAttnLayer_14800457302519
// MI455X (gfx1250) — matrix operand order repaired and checked by reading, not compiled, not run on hardware
//
#include <hip/hip_runtime.h>
#include <hip/hip_bf16.h>

// ---------- types ----------
typedef __attribute__((ext_vector_type(16))) __bf16 v16bf;
typedef __attribute__((ext_vector_type(8)))  float  v8f;
typedef __attribute__((ext_vector_type(2)))  __bf16 bf16x2;

#define E_DIM 256
#define H_DIM 8
#define D_DIM 32
#define H2_DIM 512

// ---------- f32 -> bf16 elementwise convert ----------
__global__ void cvt_bf16_kernel(const float* __restrict__ in,
                                __bf16* __restrict__ out, int n) {
    int i = blockIdx.x * blockDim.x + threadIdx.x;
    int stride = gridDim.x * blockDim.x;
    for (; i < n; i += stride) out[i] = (__bf16)in[i];
}

// ---------- f32 KxN -> bf16 NxK (transpose + convert), for B operands ----------
__global__ void cvt_tr_bf16_kernel(const float* __restrict__ in,
                                   __bf16* __restrict__ out, int K, int N) {
    int i = blockIdx.x * blockDim.x + threadIdx.x;
    if (i >= K * N) return;
    int k = i / N, n = i % N;
    out[(size_t)n * K + k] = (__bf16)in[i];
}

// ---------- bf16 WMMA GEMM: out[M,N] = g(A[M,K] @ Wt[N,K]^T + bias) ----------
// Wt stored transposed (N x K) so B-fragment K-pairs are contiguous dwords.
// 8 waves/block; each wave register-blocks a 2x2 grid of 16x16 tiles (32x32),
// block covers 64 rows x 128 cols. 4 WMMAs per 8 b128 loads per K-step.
__global__ void gemm_bf16_wmma_kernel(const __bf16* __restrict__ A,
                                      const __bf16* __restrict__ Wt,
                                      const float* __restrict__ bias,
                                      __bf16* __restrict__ out,
                                      int M, int K, int N,
                                      int relu, float scale) {
    const int lane = threadIdx.x & 31;
    const int wave = threadIdx.x >> 5;
    const int wn = wave & 3;                  // 4 n-wave-slots (2 tiles each)
    const int wm = wave >> 2;                 // 2 m-wave-slots (2 tiles each)
    const int tiles_m = M >> 4;
    const int mt0 = blockIdx.y * 4 + wm * 2;  // first m-tile of this wave
    const int nt0 = blockIdx.x * 8 + wn * 2;  // first n-tile of this wave
    if (mt0 >= tiles_m) return;               // uniform per wave: EXEC all-ones
    const bool m1ok = (mt0 + 1) < tiles_m;

    const int l16  = lane & 15;
    const int half = lane >> 4;
    const __bf16* arow0 = A + (size_t)(mt0 * 16 + l16) * K;
    const __bf16* arow1 = m1ok ? (A + (size_t)((mt0 + 1) * 16 + l16) * K) : arow0;
    const __bf16* brow0 = Wt + (size_t)(nt0 * 16 + l16) * K;
    const __bf16* brow1 = Wt + (size_t)((nt0 + 1) * 16 + l16) * K;

    v8f c00 = {}, c01 = {}, c10 = {}, c11 = {};
    for (int k0 = 0; k0 < K; k0 += 32) {
        v16bf a0, a1, b0, b1;
        // ISA 7.12.2 16-bit A 16x32: VGPR v<4 -> K = half*8 + 2v (+1)
        //                            VGPR v>=4 -> K = 16 + half*8 + 2(v-4) (+1)
        // B 32x16: same K mapping as A, col = l16
#pragma unroll
        for (int v = 0; v < 8; ++v) {
            const int ka = ((v < 4) ? 0 : 16) + half * 8 + (v & 3) * 2;
            const int kb = ((v < 4) ? 0 : 16) + half * 8 + (v & 3) * 2;
            bf16x2 p;
            p = *(const bf16x2*)(arow0 + k0 + ka); a0[2*v] = p.x; a0[2*v+1] = p.y;
            p = *(const bf16x2*)(arow1 + k0 + ka); a1[2*v] = p.x; a1[2*v+1] = p.y;
            p = *(const bf16x2*)(brow0 + k0 + kb); b0[2*v] = p.x; b0[2*v+1] = p.y;
            p = *(const bf16x2*)(brow1 + k0 + kb); b1[2*v] = p.x; b1[2*v+1] = p.y;
        }
        // (neg_a, A, neg_b, B, c_mod, C, reuse_a, reuse_b)
        c00 = __builtin_amdgcn_wmma_f32_16x16x32_bf16(false, a0, false, b0, (short)0, c00, false, false);
        c01 = __builtin_amdgcn_wmma_f32_16x16x32_bf16(false, a0, false, b1, (short)0, c01, false, false);
        c10 = __builtin_amdgcn_wmma_f32_16x16x32_bf16(false, a1, false, b0, (short)0, c10, false, false);
        c11 = __builtin_amdgcn_wmma_f32_16x16x32_bf16(false, a1, false, b1, (short)0, c11, false, false);
    }

    const int col0 = nt0 * 16 + l16;
    const int col1 = (nt0 + 1) * 16 + l16;
    const float bv0 = bias[col0];
    const float bv1 = bias[col1];
    auto store_tile = [&](const v8f& c, int mt, int col, float bv) {
#pragma unroll
        for (int r = 0; r < 8; ++r) {         // C/D: VGPR r -> row M = half*8 + r
            int m = mt * 16 + half * 8 + r;
            float val = c[r] + bv;
            if (relu) val = fmaxf(val, 0.0f);
            val *= scale;
            out[(size_t)m * N + col] = (__bf16)val;
        }
    };
    store_tile(c00, mt0, col0, bv0);
    store_tile(c01, mt0, col1, bv1);
    if (m1ok) {
        store_tile(c10, mt0 + 1, col0, bv0);
        store_tile(c11, mt0 + 1, col1, bv1);
    }
}

// ---------- attention: one wave per (t, h) ----------
__global__ void depth_attn_kernel(const __bf16* __restrict__ qf,   // T x E (pre-scaled)
                                  const __bf16* __restrict__ kf,   // S x E
                                  const float*  __restrict__ vf,   // S x E
                                  const int* __restrict__ ranks_feat,
                                  const int* __restrict__ starts,
                                  const int* __restrict__ lens,
                                  float* __restrict__ out, int T) {
    const int lane = threadIdx.x & 31;
    const int wave = threadIdx.x >> 5;
    const int task = blockIdx.x * 8 + wave;
    const int t = task >> 3;
    const int h = task & 7;
    if (t >= T) return;

    const int L  = lens[t];
    const int st = starts[t];
    const bool active = lane < L;                 // lane plays role "j"
    const int sj = active ? ranks_feat[st + lane] : 0;

    // q row: lane d holds q[t,h,d]; broadcast via shuffles (full EXEC).
    const float qd = (float)qf[(size_t)t * E_DIM + h * D_DIM + lane];

    // w_j = sum_d q[d] * k[s_j, h, d]  (lane j; all lanes execute the loop)
    const __bf16* krow = kf + (size_t)sj * E_DIM + h * D_DIM;
    float acc = 0.0f;
#pragma unroll
    for (int d = 0; d < 32; ++d)
        acc += __shfl(qd, d, 32) * (float)krow[d];
    float w = active ? acc : -3.0e38f;

    // softmax over the wave (empty slots contribute 0, matching -1e9 fill)
    float m = w;
#pragma unroll
    for (int off = 16; off; off >>= 1)
        m = fmaxf(m, __shfl_xor(m, off, 32));
    float e = active ? __expf(w - m) : 0.0f;
    float s = e;
#pragma unroll
    for (int off = 16; off; off >>= 1)
        s += __shfl_xor(s, off, 32);
    float p = e / s;

    // out[t,h,d] = sum_j p_j * v[s_j,h,d]  (lane now plays role "d"; coalesced v)
    float o = 0.0f;
    for (int j = 0; j < L; ++j) {
        float pj = __shfl(p, j, 32);
        int   si = __shfl(sj, j, 32);
        o += pj * vf[(size_t)si * E_DIM + h * D_DIM + lane];
    }
    out[(size_t)t * E_DIM + h * D_DIM + lane] = o;
}

// ---------- launch ----------
extern "C" void kernel_launch(void* const* d_in, const int* in_sizes, int n_in,
                              void* d_out, int out_size, void* d_ws, size_t ws_size,
                              hipStream_t stream) {
    (void)n_in; (void)out_size; (void)ws_size;
    const float* query_depth = (const float*)d_in[0];
    const float* key         = (const float*)d_in[1];
    const float* value       = (const float*)d_in[2];
    const float* q_w1 = (const float*)d_in[3];
    const float* q_b1 = (const float*)d_in[4];
    const float* q_w2 = (const float*)d_in[5];
    const float* q_b2 = (const float*)d_in[6];
    const float* k_w1 = (const float*)d_in[7];
    const float* k_b1 = (const float*)d_in[8];
    const float* k_w2 = (const float*)d_in[9];
    const float* k_b2 = (const float*)d_in[10];
    const int* ranks_feat = (const int*)d_in[11];
    const int* starts     = (const int*)d_in[13];
    const int* lens       = (const int*)d_in[14];
    float* out = (float*)d_out;

    const int E = E_DIM, H2 = H2_DIM;
    const int T = in_sizes[0] / E;   // 32400
    const int S = in_sizes[1] / E;   // 16896

    // carve workspace (bf16 buffers, 256B aligned): ~85 MB total
    char* ws = (char*)d_ws;
    size_t off = 0;
    auto carve = [&](size_t elems) {
        __bf16* p = (__bf16*)(ws + off);
        off += ((elems * 2 + 255) / 256) * 256;
        return p;
    };
    __bf16* qW1t   = carve((size_t)E * H2);
    __bf16* qW2t   = carve((size_t)H2 * E);
    __bf16* kW1t   = carve((size_t)E * H2);
    __bf16* kW2t   = carve((size_t)H2 * E);
    __bf16* qd_bf  = carve((size_t)T * E);
    __bf16* key_bf = carve((size_t)S * E);
    __bf16* hid_bf = carve((size_t)(T > S ? T : S) * H2);
    __bf16* qo_bf  = carve((size_t)T * E);
    __bf16* ko_bf  = carve((size_t)S * E);

    // 1) weights: transpose + convert to bf16 (N x K layout)
    {
        int n = E * H2, g = (n + 255) / 256;
        cvt_tr_bf16_kernel<<<g, 256, 0, stream>>>(q_w1, qW1t, E,  H2);
        cvt_tr_bf16_kernel<<<g, 256, 0, stream>>>(q_w2, qW2t, H2, E);
        cvt_tr_bf16_kernel<<<g, 256, 0, stream>>>(k_w1, kW1t, E,  H2);
        cvt_tr_bf16_kernel<<<g, 256, 0, stream>>>(k_w2, kW2t, H2, E);
    }
    // 2) activations -> bf16
    cvt_bf16_kernel<<<2048, 256, 0, stream>>>(query_depth, qd_bf, T * E);
    cvt_bf16_kernel<<<2048, 256, 0, stream>>>(key,         key_bf, S * E);

    // 3) MLPs via WMMA GEMMs (grid.x = N/128, grid.y = ceil((M/16)/4))
    const float qscale = 0.17677669529663687f;  // 1/sqrt(D=32)
    const int gyT = (T / 16 + 3) / 4;           // 507
    const int gyS = (S / 16 + 3) / 4;           // 264
    gemm_bf16_wmma_kernel<<<dim3(H2 / 128, gyT), 256, 0, stream>>>(
        qd_bf, qW1t, q_b1, hid_bf, T, E, H2, 1, 1.0f);
    gemm_bf16_wmma_kernel<<<dim3(E / 128, gyT), 256, 0, stream>>>(
        hid_bf, qW2t, q_b2, qo_bf, T, H2, E, 0, qscale);
    gemm_bf16_wmma_kernel<<<dim3(H2 / 128, gyS), 256, 0, stream>>>(
        key_bf, kW1t, k_b1, hid_bf, S, E, H2, 1, 1.0f);
    gemm_bf16_wmma_kernel<<<dim3(E / 128, gyS), 256, 0, stream>>>(
        hid_bf, kW2t, k_b2, ko_bf, S, H2, E, 0, 1.0f);

    // 4) gathered softmax-attention: one wave per (t,h); T blocks x 8 waves
    depth_attn_kernel<<<T, 256, 0, stream>>>(
        qo_bf, ko_bf, value, ranks_feat, starts, lens, out, T);
}
